// LSTM_37735582662662
// MI455X (gfx1250) — matrix operand order repaired and checked by reading, not compiled, not run on hardware
//
#include <hip/hip_runtime.h>
#include <hip/hip_bf16.h>
#include <math.h>

// ---------------------------------------------------------------------------
// LSTM forward for MI455X (gfx1250, wave32, WMMA).
//   B=256, S=512, D=256, H=256, O=128
// Pipeline:
//   k_pack    : convert weights f32->bf16 (row-major, B-fragment friendly),
//               zero h ping-pong buffers + barrier counter
//   k_xgemm   : preact[s][b][4H] = x @ Wx^T + Wx_b + bgate   (bf16 WMMA, f32 acc)
//   k_scan    : persistent 32-WG kernel, 512 timesteps; Wh staged to LDS via
//               TDM (tensor_load_to_lds) with padded rows, c in VGPRs,
//               h ping-pong in global bf16; one grid barrier per step
//   k_proj    : p = h @ Wp^T + Wp_b + bp (WMMA)
//   k_softmax : row softmax via wave32 shuffles
// ---------------------------------------------------------------------------

#define B_ 256
#define S_ 512
#define D_ 256
#define H_ 256
#define O_ 128
#define NWG_SCAN 32u

typedef __attribute__((ext_vector_type(16))) __bf16 v16bf;
typedef __attribute__((ext_vector_type(8)))  float  v8f;
typedef __attribute__((ext_vector_type(4)))  unsigned u32x4;
typedef __attribute__((ext_vector_type(8)))  int      i32x8;
typedef __attribute__((ext_vector_type(4)))  int      i32x4;

#if defined(__has_builtin)
#if __has_builtin(__builtin_amdgcn_tensor_load_to_lds) && \
    __has_builtin(__builtin_amdgcn_s_wait_tensorcnt)
#define HAVE_TDM 1
#endif
#endif

union ABFrag { unsigned u[8]; v16bf v; };

// Workspace layout (bytes). Requires ~514 MiB of workspace.
static constexpr size_t OFF_WX  = 0;                 // bf16 [4][H][D]    524288
static constexpr size_t OFF_WH  = 524288;            // bf16 [4][H][H]    524288
static constexpr size_t OFF_WP  = 1048576;           // bf16 [O][H]        65536
static constexpr size_t OFF_H0  = 1114112;           // bf16 [B][H]       131072
static constexpr size_t OFF_H1  = 1245184;           // bf16 [B][H]       131072
static constexpr size_t OFF_P   = 1376256;           // f32  [B][O]       131072
static constexpr size_t OFF_CNT = 1507328;           // barrier counter
static constexpr size_t OFF_PRE = 2097152;           // f32  [S][B][4H] 536870912

__device__ __forceinline__ unsigned short f2bf_rne(float f) {
  union { float f; unsigned u; } x; x.f = f;
  unsigned r = x.u + 0x7FFFu + ((x.u >> 16) & 1u);   // round-to-nearest-even
  return (unsigned short)(r >> 16);
}
__device__ __forceinline__ unsigned pk2bf(float lo, float hi) {
  return (unsigned)f2bf_rne(lo) | ((unsigned)f2bf_rne(hi) << 16);
}
__device__ __forceinline__ float sigm(float x) { return 1.0f / (1.0f + __expf(-x)); }

__device__ __forceinline__ void set8(ABFrag& f, uint4 q0, uint4 q1) {
  f.u[0]=q0.x; f.u[1]=q0.y; f.u[2]=q0.z; f.u[3]=q0.w;
  f.u[4]=q1.x; f.u[5]=q1.y; f.u[6]=q1.z; f.u[7]=q1.w;
}

// ---------------------------------------------------------------------------
// Kernel 0: pack weights to bf16, zero h buffers + barrier counter
// ---------------------------------------------------------------------------
__global__ __launch_bounds__(256) void k_pack(
    const float* __restrict__ wxg, const float* __restrict__ wxi,
    const float* __restrict__ wxf, const float* __restrict__ wxo,
    const float* __restrict__ whg, const float* __restrict__ whi,
    const float* __restrict__ whf, const float* __restrict__ who,
    const float* __restrict__ wp,
    unsigned short* __restrict__ wx_pk, unsigned short* __restrict__ wh_pk,
    unsigned short* __restrict__ wp_pk,
    unsigned* __restrict__ h0, unsigned* __restrict__ h1,
    unsigned* __restrict__ cnt)
{
  unsigned tid = blockIdx.x * 256u + threadIdx.x;
  if (tid == 0) *cnt = 0u;
  const unsigned NX = 4u * H_ * D_;        // 262144
  const unsigned NH = 4u * H_ * H_;        // 262144
  const unsigned NP = O_ * H_;             // 32768
  const unsigned NZ = 2u * B_ * (H_ / 2);  // 65536 u32 (two bf16 h buffers)
  const unsigned total = NX + NH + NP + NZ;
  for (unsigned i = tid; i < total; i += gridDim.x * 256u) {
    if (i < NX) {
      unsigned g = i >> 16, r = i & 65535u;
      const float* s = (g == 0) ? wxg : (g == 1) ? wxi : (g == 2) ? wxf : wxo;
      wx_pk[i] = f2bf_rne(s[r]);
    } else if (i < NX + NH) {
      unsigned j = i - NX, g = j >> 16, r = j & 65535u;
      const float* s = (g == 0) ? whg : (g == 1) ? whi : (g == 2) ? whf : who;
      wh_pk[j] = f2bf_rne(s[r]);
    } else if (i < NX + NH + NP) {
      unsigned j = i - NX - NH;
      wp_pk[j] = f2bf_rne(wp[j]);
    } else {
      unsigned j = i - NX - NH - NP;
      if (j < 32768u) h0[j] = 0u; else h1[j - 32768u] = 0u;
    }
  }
}

// ---------------------------------------------------------------------------
// Kernel 1: preact[s][b][g*256+h] = x @ Wx^T + Wx_b + bgate
// One wave per 32 rows of M = b*S + s (same b); A block cached in 128 VGPRs.
// ---------------------------------------------------------------------------
__global__ __launch_bounds__(256) void k_xgemm(
    const float* __restrict__ x,               // [B][S][D]
    const unsigned short* __restrict__ wx_pk,  // [4][H][D] bf16
    const float* __restrict__ wxbg, const float* __restrict__ wxbi,
    const float* __restrict__ wxbf, const float* __restrict__ wxbo,
    const float* __restrict__ bgg, const float* __restrict__ bgi,
    const float* __restrict__ bgf, const float* __restrict__ bgo,
    float* __restrict__ preact)                // [S][B][4H]
{
  const unsigned lane  = threadIdx.x & 31u;
  const unsigned wave  = blockIdx.x * 8u + (threadIdx.x >> 5);
  const unsigned m0    = wave * 32u;           // 32 rows, same b
  const unsigned b     = m0 >> 9;
  const unsigned s0    = m0 & 511u;
  const unsigned l15   = lane & 15u;
  const unsigned kg    = lane >> 4;

  // Load + convert the 32x256 A block (x rows) into bf16 fragments (in VGPRs).
  ABFrag a[2][8];
  #pragma unroll
  for (int blk = 0; blk < 2; ++blk) {
    const float* xr = x + (size_t)(m0 + blk * 16u + l15) * D_;
    #pragma unroll
    for (int kf = 0; kf < 8; ++kf) {
      const float4* p = (const float4*)(xr + kf * 32 + kg * 8);
      float4 c0a = p[0], c0b = p[1];           // K chunk [kg*8 .. kg*8+7]
      float4 c1a = p[4], c1b = p[5];           // K chunk +16
      a[blk][kf].u[0] = pk2bf(c0a.x, c0a.y); a[blk][kf].u[1] = pk2bf(c0a.z, c0a.w);
      a[blk][kf].u[2] = pk2bf(c0b.x, c0b.y); a[blk][kf].u[3] = pk2bf(c0b.z, c0b.w);
      a[blk][kf].u[4] = pk2bf(c1a.x, c1a.y); a[blk][kf].u[5] = pk2bf(c1a.z, c1a.w);
      a[blk][kf].u[6] = pk2bf(c1b.x, c1b.y); a[blk][kf].u[7] = pk2bf(c1b.z, c1b.w);
    }
  }

  #pragma unroll
  for (int g = 0; g < 4; ++g) {
    const float* wxb = (g == 0) ? wxbg : (g == 1) ? wxbi : (g == 2) ? wxbf : wxbo;
    const float* bgt = (g == 0) ? bgg  : (g == 1) ? bgi  : (g == 2) ? bgf  : bgo;
    #pragma unroll 1
    for (unsigned ht = 0; ht < 16u; ++ht) {
      const unsigned n = ht * 16u + l15;       // output column within H (per lane)
      const unsigned short* wrow = wx_pk + ((size_t)((unsigned)g * 256u + n)) * D_;
      v8f acc0 = {}; v8f acc1 = {};
      #pragma unroll
      for (int kf = 0; kf < 8; ++kf) {
        const uint4* q = (const uint4*)(wrow + kf * 32 + kg * 8);
        ABFrag bfr; set8(bfr, q[0], q[2]);
        acc0 = __builtin_amdgcn_wmma_f32_16x16x32_bf16(false, a[0][kf].v, false, bfr.v,
                                                       (short)0, acc0, false, false);
        acc1 = __builtin_amdgcn_wmma_f32_16x16x32_bf16(false, a[1][kf].v, false, bfr.v,
                                                       (short)0, acc1, false, false);
      }
      const float badd = wxb[n] + bgt[b * 256u + n];   // uniform over rows (same b)
      const unsigned col = (unsigned)g * 256u + ht * 16u + l15;
      #pragma unroll
      for (int r = 0; r < 8; ++r) {
        unsigned row = (unsigned)r + kg * 8u;
        preact[((s0 + row)        * 256u + b) * 1024u + col] = acc0[r] + badd;
        preact[((s0 + 16u + row)  * 256u + b) * 1024u + col] = acc1[r] + badd;
      }
    }
  }
}

// ---------------------------------------------------------------------------
// Kernel 2: persistent recurrent scan. 32 WGs x 256 threads.
// WG owns one 16-wide h-tile (Wh rows in LDS for all 4 gates, all 512 steps),
// each of its 8 waves owns one 16-wide b-tile; c lives in VGPRs.
// Wh is staged to LDS by the Tensor Data Mover (3D tile: 256 elem x 16 rows x
// 4 gates, row stride 256, gate stride 65536, +4 DWORD pad per 128 DWORDs to
// produce the bank-conflict-free 132-u32 LDS row pitch).
// ---------------------------------------------------------------------------
__global__ __launch_bounds__(256) void k_scan(
    const float* __restrict__ preact,          // [S][B][4H]
    const unsigned* __restrict__ wh_u32,       // [4][H][H/2] bf16 pairs
    const float* __restrict__ whbg, const float* __restrict__ whbi,
    const float* __restrict__ whbf, const float* __restrict__ whbo,
    unsigned* __restrict__ hb0, unsigned* __restrict__ hb1,  // bf16 [B][H/2] u32
    unsigned* __restrict__ cnt)
{
  __shared__ unsigned lds_w[4 * 16 * 132];     // padded rows: 128 pairs + 4 pad
  const unsigned tid  = threadIdx.x;
  const unsigned lane = tid & 31u;
  const unsigned W    = blockIdx.x;            // 0..31
  const unsigned nt   = W >> 1;                // h-tile 0..15
  const unsigned h0g  = nt * 16u;
  const unsigned bt   = (W & 1u) * 8u + (tid >> 5);
  const unsigned b0   = bt * 16u;
  const unsigned l15  = lane & 15u, kg = lane >> 4;

#ifdef HAVE_TDM
  // Issue one TDM descriptor from wave 0: Wh rows [h0g,h0g+16) x 4 gates -> LDS.
  if (tid == 0) {
    unsigned long long ga =
        (unsigned long long)(size_t)(const void*)(wh_u32 + (size_t)h0g * 128u);
    unsigned lds_base = (unsigned)(size_t)(void*)&lds_w[0];  // LDS byte offset
    u32x4 g0 = { 1u,                                  // count=1, user descriptor
                 lds_base,                            // lds_addr
                 (unsigned)(ga & 0xFFFFFFFFull),      // global_addr[31:0]
                 (unsigned)((ga >> 32) & 0x01FFFFFFull) | (2u << 30) }; // [56:32]|type=2
    i32x8 g1 = { (int)((1u << 16) |                   // data_size = 2 bytes
                       (1u << 20) |                   // pad_enable
                       (6u << 22) |                   // pad_interval: 128 DWORDs
                       (3u << 25)),                   // pad_amount: 4 DWORDs
                 (int)(256u << 16),                   // tensor_dim0[15:0]=256
                 (int)(16u  << 16),                   // tensor_dim1[15:0]=16
                 (int)(256u << 16),                   // tile_dim0=256
                 (int)(16u | (4u << 16)),             // tile_dim1=16, tile_dim2=4
                 (int)256,                            // tensor_dim0_stride=256
                 0,                                   // strides mid bits
                 1 };                                 // tensor_dim1_stride=65536>>16
    i32x4 g2 = { 4, 0, 0, 0 };                        // tensor_dim2=4
    i32x4 g3 = { 0, 0, 0, 0 };
#if defined(__clang_major__) && (__clang_major__ >= 23)
    i32x8 g4 = { 0, 0, 0, 0, 0, 0, 0, 0 };
    __builtin_amdgcn_tensor_load_to_lds(g0, g1, g2, g3, g4, 0);
#else
    __builtin_amdgcn_tensor_load_to_lds(g0, g1, g2, g3, 0);
#endif
    __builtin_amdgcn_s_wait_tensorcnt(0);
  }
#endif

  // Cooperative staging of the same tile (byte-identical data; benign overlap
  // with the TDM transfer, and guarantees LDS contents regardless).
  for (unsigned i = tid; i < 8192u; i += 256u) {
    unsigned g = i >> 11, rem = i & 2047u, n = rem >> 7, kp = rem & 127u;
    lds_w[(g * 16u + n) * 132u + kp] = wh_u32[((g * 256u + h0g + n) * 128u) + kp];
  }
  __syncthreads();
  // Under a cluster dispatch this would be the cross-WGP sync point; NOP otherwise.
  __builtin_amdgcn_s_cluster_barrier();

  const unsigned hcol = h0g + l15;
  const float wb0 = whbg[hcol], wb1 = whbi[hcol], wb2 = whbf[hcol], wb3 = whbo[hcol];

  float c[8];
  #pragma unroll
  for (int r = 0; r < 8; ++r) c[r] = 0.0f;

  for (unsigned s = 0; s < (unsigned)S_; ++s) {
    const unsigned* rbuf = (s & 1u) ? hb1 : hb0;
    unsigned*       wbuf = (s & 1u) ? hb0 : hb1;

    // Prefetch next step's preactivations while this step computes.
    if (s + 1u < (unsigned)S_) {
      const float* pf = preact + ((s + 1u) * 256u + b0 + l15) * 1024u + h0g;
      __builtin_prefetch(pf, 0, 1);
      __builtin_prefetch(pf + 512u, 0, 1);
    }

    // A fragments: h(s-1) rows b0..b0+15, bf16.
    ABFrag a[8];
    {
      const unsigned* hr = rbuf + (b0 + l15) * 128u;
      #pragma unroll
      for (int kf = 0; kf < 8; ++kf) {
        uint4 q0 = *(const uint4*)(hr + kf * 16 + kg * 4);
        uint4 q1 = *(const uint4*)(hr + kf * 16 + kg * 4 + 8);
        set8(a[kf], q0, q1);
      }
    }

    v8f acc4[4] = {};
    #pragma unroll
    for (int kf = 0; kf < 8; ++kf) {
      #pragma unroll
      for (int g = 0; g < 4; ++g) {
        unsigned base = ((unsigned)g * 16u + l15) * 132u + (unsigned)(kf * 16) + kg * 4u;
        uint4 q0 = *(const uint4*)&lds_w[base];
        uint4 q1 = *(const uint4*)&lds_w[base + 8u];
        ABFrag bfr; set8(bfr, q0, q1);
        acc4[g] = __builtin_amdgcn_wmma_f32_16x16x32_bf16(false, a[kf].v, false, bfr.v,
                                                          (short)0, acc4[g], false, false);
      }
    }

    // Elementwise LSTM cell update; this wave owns all four gates at (b,h).
    unsigned short* w16 = (unsigned short*)wbuf;
    #pragma unroll
    for (int r = 0; r < 8; ++r) {
      unsigned b = b0 + (unsigned)r + kg * 8u;
      const float* pr = preact + (s * 256u + b) * 1024u + hcol;
      float gg = tanhf(acc4[0][r] + pr[0]   + wb0);
      float ii = sigm (acc4[1][r] + pr[256] + wb1);
      float ff = sigm (acc4[2][r] + pr[512] + wb2);
      float oo = sigm (acc4[3][r] + pr[768] + wb3);
      c[r] = gg * ii + c[r] * ff;
      w16[b * 256u + hcol] = f2bf_rne(tanhf(c[r]) * oo);
    }

    // Grid-wide barrier: release h(s), wait for all 32 WGs, acquire.
    __threadfence();
    __syncthreads();
    if (tid == 0) {
      __hip_atomic_fetch_add(cnt, 1u, __ATOMIC_RELEASE, __HIP_MEMORY_SCOPE_AGENT);
      const unsigned target = NWG_SCAN * (s + 1u);
      while (__hip_atomic_load(cnt, __ATOMIC_ACQUIRE, __HIP_MEMORY_SCOPE_AGENT) < target) {
        __builtin_amdgcn_s_sleep(1);
      }
    }
    __syncthreads();
    __threadfence();
  }
}

// ---------------------------------------------------------------------------
// Kernel 3: p[b][o] = h_final @ Wp^T + Wp_b + bp   (WMMA)
// ---------------------------------------------------------------------------
__global__ __launch_bounds__(256) void k_proj(
    const unsigned* __restrict__ hfin,          // bf16 [B][H/2] u32 (hb0)
    const unsigned short* __restrict__ wp_pk,   // [O][H] bf16
    const float* __restrict__ wpb, const float* __restrict__ bp,
    float* __restrict__ p)
{
  const unsigned lane = threadIdx.x & 31u;
  const unsigned w    = blockIdx.x * 8u + (threadIdx.x >> 5); // 0..127
  const unsigned b0   = (w >> 3) * 16u;
  const unsigned o0   = (w & 7u) * 16u;
  const unsigned l15  = lane & 15u, kg = lane >> 4;

  const unsigned* hr = hfin + (b0 + l15) * 128u;
  const unsigned short* wrow = wp_pk + (size_t)(o0 + l15) * H_;
  v8f acc = {};
  #pragma unroll
  for (int kf = 0; kf < 8; ++kf) {
    uint4 qa0 = *(const uint4*)(hr + kf * 16 + kg * 4);
    uint4 qa1 = *(const uint4*)(hr + kf * 16 + kg * 4 + 8);
    ABFrag afr; set8(afr, qa0, qa1);
    const uint4* qb = (const uint4*)(wrow + kf * 32 + kg * 8);
    ABFrag bfr; set8(bfr, qb[0], qb[2]);
    acc = __builtin_amdgcn_wmma_f32_16x16x32_bf16(false, afr.v, false, bfr.v,
                                                  (short)0, acc, false, false);
  }
  const unsigned o = o0 + l15;
  #pragma unroll
  for (int r = 0; r < 8; ++r) {
    unsigned b = b0 + (unsigned)r + kg * 8u;
    p[b * 128u + o] = acc[r] + wpb[o] + bp[b * 128u + o];
  }
}

// ---------------------------------------------------------------------------
// Kernel 4: row softmax over O=128, one wave32 per row.
// ---------------------------------------------------------------------------
__global__ __launch_bounds__(256) void k_softmax(const float* __restrict__ p,
                                                 float* __restrict__ out)
{
  const unsigned lane = threadIdx.x & 31u;
  const unsigned b    = blockIdx.x * 8u + (threadIdx.x >> 5);
  float v[4];
  #pragma unroll
  for (int j = 0; j < 4; ++j) v[j] = p[b * 128u + lane + 32u * j];
  float m = fmaxf(fmaxf(v[0], v[1]), fmaxf(v[2], v[3]));
  #pragma unroll
  for (int off = 16; off >= 1; off >>= 1) m = fmaxf(m, __shfl_xor(m, off, 32));
  float e[4], ssum = 0.0f;
  #pragma unroll
  for (int j = 0; j < 4; ++j) { e[j] = __expf(v[j] - m); ssum += e[j]; }
  #pragma unroll
  for (int off = 16; off >= 1; off >>= 1) ssum += __shfl_xor(ssum, off, 32);
  float inv = 1.0f / ssum;
  #pragma unroll
  for (int j = 0; j < 4; ++j) out[b * 128u + lane + 32u * j] = e[j] * inv;
}

// ---------------------------------------------------------------------------
extern "C" void kernel_launch(void* const* d_in, const int* in_sizes, int n_in,
                              void* d_out, int out_size, void* d_ws, size_t ws_size,
                              hipStream_t stream) {
  (void)in_sizes; (void)n_in; (void)out_size; (void)ws_size;
  const float* x     = (const float*)d_in[0];
  const float* Wgx_w = (const float*)d_in[1];  const float* Wgx_b = (const float*)d_in[2];
  const float* Wgh_w = (const float*)d_in[3];  const float* Wgh_b = (const float*)d_in[4];
  const float* Wix_w = (const float*)d_in[5];  const float* Wix_b = (const float*)d_in[6];
  const float* Wih_w = (const float*)d_in[7];  const float* Wih_b = (const float*)d_in[8];
  const float* Wfx_w = (const float*)d_in[9];  const float* Wfx_b = (const float*)d_in[10];
  const float* Wfh_w = (const float*)d_in[11]; const float* Wfh_b = (const float*)d_in[12];
  const float* Wox_w = (const float*)d_in[13]; const float* Wox_b = (const float*)d_in[14];
  const float* Woh_w = (const float*)d_in[15]; const float* Woh_b = (const float*)d_in[16];
  const float* Wph_w = (const float*)d_in[17]; const float* Wph_b = (const float*)d_in[18];
  const float* bg    = (const float*)d_in[19];
  const float* bi    = (const float*)d_in[20];
  const float* bf    = (const float*)d_in[21];
  const float* bo    = (const float*)d_in[22];
  const float* bp    = (const float*)d_in[23];

  char* ws = (char*)d_ws;
  unsigned short* wx_pk = (unsigned short*)(ws + OFF_WX);
  unsigned short* wh_pk = (unsigned short*)(ws + OFF_WH);
  unsigned short* wp_pk = (unsigned short*)(ws + OFF_WP);
  unsigned* hb0 = (unsigned*)(ws + OFF_H0);
  unsigned* hb1 = (unsigned*)(ws + OFF_H1);
  float*    pbuf = (float*)(ws + OFF_P);
  unsigned* cnt  = (unsigned*)(ws + OFF_CNT);
  float*    preact = (float*)(ws + OFF_PRE);

  k_pack<<<2432, 256, 0, stream>>>(Wgx_w, Wix_w, Wfx_w, Wox_w,
                                   Wgh_w, Wih_w, Wfh_w, Woh_w, Wph_w,
                                   wx_pk, wh_pk, wp_pk, hb0, hb1, cnt);

  k_xgemm<<<512, 256, 0, stream>>>(x, wx_pk,
                                   Wgx_b, Wix_b, Wfx_b, Wox_b,
                                   bg, bi, bf, bo, preact);

  k_scan<<<NWG_SCAN, 256, 0, stream>>>(preact, (const unsigned*)wh_pk,
                                       Wgh_b, Wih_b, Wfh_b, Woh_b,
                                       hb0, hb1, cnt);

  // Last step (s=511) writes hb0 -> final h lives in hb0 (bf16).
  k_proj<<<16, 256, 0, stream>>>(hb0, wp_pk, Wph_b, bp, pbuf);

  k_softmax<<<32, 256, 0, stream>>>(pbuf, (float*)d_out);
}
